// MultiHeadAttentionBlock_74878459838597
// MI455X (gfx1250) — matrix operand order repaired and checked by reading, not compiled, not run on hardware
//
#include <hip/hip_runtime.h>
#include <hip/hip_bf16.h>

typedef __bf16 bf16_t;
typedef __attribute__((ext_vector_type(16))) __bf16 v16bf;
typedef __attribute__((ext_vector_type(8)))  __bf16 v8bf;
typedef __attribute__((ext_vector_type(4)))  __bf16 v4bf;
typedef __attribute__((ext_vector_type(8)))  float  v8f;

union FragBF { v16bf v; v8bf h[2]; };

static constexpr int Bb = 4, Ss = 2048, Dd = 1024, Hh = 16, HDd = 64;
static constexpr int Mtot = Bb * Ss;                 // 8192 projection rows
static constexpr float kScaleLog2e = 0.125f * 1.4426950408889634f;  // 1/sqrt(64)*log2(e)

#define WMMA_BF16(A, Bm, C) \
  __builtin_amdgcn_wmma_f32_16x16x32_bf16(false, (A), false, (Bm), (short)0, (C), false, false)

// ---------------------------------------------------------------------------
// Kernel 0: bit-pack the [S,S] int32 mask into one bit per element using
// wave32 ballot. bits[q*(S/32) + w] bit b == (mask[q][w*32+b] != 0).
// ---------------------------------------------------------------------------
__global__ __launch_bounds__(256) void mask_pack_kernel(
    const int* __restrict__ mask, unsigned* __restrict__ bits)
{
  const size_t idx = (size_t)blockIdx.x * 256 + threadIdx.x;
  const unsigned long long bal = __ballot(mask[idx] != 0);
  if ((threadIdx.x & 31) == 0) bits[idx >> 5] = (unsigned)bal;
}

// ---------------------------------------------------------------------------
// Kernel 1: fused Q/K/V projection. C = X @ W^T, X:[8192,1024] f32,
// W:[1024,1024] f32, out bf16 in head-split [B,H,S,HD] layout.
// Double-buffered LDS, software-pipelined global->LDS staging.
// ---------------------------------------------------------------------------
__global__ __launch_bounds__(256) void qkv_proj_kernel(
    const float* __restrict__ X0, const float* __restrict__ X1, const float* __restrict__ X2,
    const float* __restrict__ W0, const float* __restrict__ W1, const float* __restrict__ W2,
    bf16_t* __restrict__ O0, bf16_t* __restrict__ O1, bf16_t* __restrict__ O2)
{
  constexpr int BM = 128, BN = 128, BK = 32, LDT = BK + 8;   // 40 el = 80B rows
  constexpr int NK = Dd / BK;                                // 32 k-tiles
  __shared__ bf16_t ldsA[2][BM * LDT];
  __shared__ bf16_t ldsB[2][BN * LDT];

  const float* X; const float* W; bf16_t* O;
  if (blockIdx.z == 0)      { X = X0; W = W0; O = O0; }
  else if (blockIdx.z == 1) { X = X1; W = W1; O = O1; }
  else                      { X = X2; W = W2; O = O2; }

  const int tid  = threadIdx.x;
  const int lane = tid & 31, wave = tid >> 5;
  const int l15  = lane & 15, half = lane >> 4;
  const int waveM = wave >> 1, waveN = wave & 1;             // 4 x 2 wave grid
  const int m0 = blockIdx.y * BM;
  const int n0 = blockIdx.x * BN;
  const int ldRow = tid >> 1;                                // 0..127
  const int ldCol = (tid & 1) * 16;                          // 0 or 16

  const float* xrow = &X[(size_t)(m0 + ldRow) * Dd + ldCol];
  const float* wrow = &W[(size_t)(n0 + ldRow) * Dd + ldCol];

  float4 fa[4], fb[4];
  auto ld_global = [&](int k0) {
#pragma unroll
    for (int i = 0; i < 4; ++i) {
      fa[i] = *(const float4*)&xrow[k0 + 4 * i];
      fb[i] = *(const float4*)&wrow[k0 + 4 * i];
    }
  };
  auto st_lds = [&](int buf) {
    bf16_t* A = &ldsA[buf][ldRow * LDT + ldCol];
    bf16_t* Bt = &ldsB[buf][ldRow * LDT + ldCol];
#pragma unroll
    for (int i = 0; i < 4; ++i) {
      v4bf ba; ba[0] = (bf16_t)fa[i].x; ba[1] = (bf16_t)fa[i].y;
               ba[2] = (bf16_t)fa[i].z; ba[3] = (bf16_t)fa[i].w;
      *(v4bf*)&A[4 * i] = ba;
      v4bf bb; bb[0] = (bf16_t)fb[i].x; bb[1] = (bf16_t)fb[i].y;
               bb[2] = (bf16_t)fb[i].z; bb[3] = (bf16_t)fb[i].w;
      *(v4bf*)&Bt[4 * i] = bb;
    }
  };

  const v8f vzero = {0.f,0.f,0.f,0.f,0.f,0.f,0.f,0.f};
  v8f acc[2][4];
#pragma unroll
  for (int mi = 0; mi < 2; ++mi)
#pragma unroll
    for (int ni = 0; ni < 4; ++ni) acc[mi][ni] = vzero;

  ld_global(0);
  st_lds(0);
  __syncthreads();

  for (int kt = 0; kt < NK; ++kt) {
    const int cur = kt & 1;
    if (kt + 1 < NK) ld_global((kt + 1) * BK);        // hide latency behind WMMA
    if (kt + 2 < NK) {
      __builtin_prefetch(&xrow[(kt + 2) * BK], 0, 1); // warm L2 two tiles ahead
      __builtin_prefetch(&wrow[(kt + 2) * BK], 0, 1);
    }

    // preload all fragments so the 8 WMMAs issue back-to-back
    FragBF a[2], b[4];
#pragma unroll
    for (int mi = 0; mi < 2; ++mi) {
      const int r = waveM * 32 + mi * 16 + l15;
      a[mi].h[0] = *(const v8bf*)&ldsA[cur][r * LDT + half * 8];
      a[mi].h[1] = *(const v8bf*)&ldsA[cur][r * LDT + 16 + half * 8];
    }
#pragma unroll
    for (int ni = 0; ni < 4; ++ni) {
      const int c = waveN * 64 + ni * 16 + l15;
      b[ni].h[0] = *(const v8bf*)&ldsB[cur][c * LDT + half * 8];
      b[ni].h[1] = *(const v8bf*)&ldsB[cur][c * LDT + 16 + half * 8];
    }
#pragma unroll
    for (int ni = 0; ni < 4; ++ni)
#pragma unroll
      for (int mi = 0; mi < 2; ++mi)
        acc[mi][ni] = WMMA_BF16(a[mi].v, b[ni].v, acc[mi][ni]);

    if (kt + 1 < NK) st_lds(cur ^ 1);
    __syncthreads();
  }

  // epilogue: f32 accum -> bf16, scattered into [B,H,S,HD]
#pragma unroll
  for (int mi = 0; mi < 2; ++mi)
#pragma unroll
    for (int ni = 0; ni < 4; ++ni)
#pragma unroll
      for (int r = 0; r < 8; ++r) {
        const int gm = m0 + waveM * 32 + mi * 16 + r + 8 * half;   // b*S + s
        const int gn = n0 + waveN * 64 + ni * 16 + l15;            // h*HD + hd
        const int bi = gm >> 11, si = gm & (Ss - 1);
        const int hi = gn >> 6,  hd = gn & (HDd - 1);
        O[(((size_t)bi * Hh + hi) * Ss + si) * HDd + hd] = (bf16_t)acc[mi][ni][r];
      }
}

// ---------------------------------------------------------------------------
// Kernel 2: flash attention per (b,h). 128 threads = 4 waves, 64-row query
// tile, streams 64-row K/V tiles through double-buffered LDS. Softmax done
// in exp2 domain (v_exp_f32 is native exp2), mask read from packed bits.
// ---------------------------------------------------------------------------
__global__ __launch_bounds__(128) void attn_kernel(
    const bf16_t* __restrict__ Qh, const bf16_t* __restrict__ Kh,
    const bf16_t* __restrict__ Vh, const unsigned* __restrict__ mbits,
    bf16_t* __restrict__ Oa)
{
  constexpr int TQ = 64, TK = 64, LDK = HDd + 8;   // 72 el = 144B rows
  constexpr int NT = Ss / TK;                      // 32 key tiles
  __shared__ bf16_t ldsK[2][TK * LDK];             // [key][hd]
  __shared__ bf16_t ldsV[2][HDd * LDK];            // transposed: [hd][key]
  __shared__ bf16_t ldsP[4][16 * LDK];             // per-wave P tile [qrow][key]

  const int tid  = threadIdx.x;
  const int lane = tid & 31, wave = tid >> 5;
  const int l15  = lane & 15, half = lane >> 4;
  const int bh = blockIdx.y;                       // b*H + h
  const int b  = bh >> 4, h = bh & (Hh - 1);
  const size_t base = (size_t)bh * Ss * HDd;
  const int q0 = blockIdx.x * TQ;
  const int qw = q0 + wave * 16;

  // Q fragments held in registers for the whole kernel (HD=64 -> 2 k-steps)
  FragBF qa[2];
  {
    const bf16_t* qp = &Qh[base + (size_t)(qw + l15) * HDd];
#pragma unroll
    for (int kk = 0; kk < 2; ++kk) {
      qa[kk].h[0] = *(const v8bf*)&qp[kk * 32 + half * 8];
      qa[kk].h[1] = *(const v8bf*)&qp[kk * 32 + 16 + half * 8];
    }
  }

  const int ldRow = tid >> 1;                      // key row 0..63
  const int ldCol = (tid & 1) * 32;                // 0 or 32
  const bf16_t* kbase_p = &Kh[base + (size_t)ldRow * HDd + ldCol];
  const bf16_t* vbase_p = &Vh[base + (size_t)ldRow * HDd + ldCol];

  v8bf kreg[4], vreg[4];
  auto ld_tiles = [&](int kt) {
    const size_t off = (size_t)kt * TK * HDd;
#pragma unroll
    for (int j = 0; j < 4; ++j) {
      kreg[j] = *(const v8bf*)&kbase_p[off + 8 * j];
      vreg[j] = *(const v8bf*)&vbase_p[off + 8 * j];
    }
  };
  auto st_tiles = [&](int buf) {
#pragma unroll
    for (int j = 0; j < 4; ++j) {
      *(v8bf*)&ldsK[buf][ldRow * LDK + ldCol + 8 * j] = kreg[j];
#pragma unroll
      for (int e = 0; e < 8; ++e)                  // transpose V into [hd][key]
        ldsV[buf][(ldCol + 8 * j + e) * LDK + ldRow] = vreg[j][e];
    }
  };

  const v8f vzero = {0.f,0.f,0.f,0.f,0.f,0.f,0.f,0.f};
  v8f o_acc[4] = {vzero, vzero, vzero, vzero};
  float m_i[8], l_i[8];
#pragma unroll
  for (int r = 0; r < 8; ++r) { m_i[r] = -1e30f; l_i[r] = 0.f; }

  ld_tiles(0);
  st_tiles(0);
  __syncthreads();

  for (int kt = 0; kt < NT; ++kt) {
    const int cur = kt & 1;
    if (kt + 1 < NT) ld_tiles(kt + 1);             // pipeline next K/V tile

    // S = Q @ K^T  (16x64 per wave = 4 accums); preload B frags per nt
    v8f s_acc[4] = {vzero, vzero, vzero, vzero};
#pragma unroll
    for (int nt = 0; nt < 4; ++nt) {
#pragma unroll
      for (int kk = 0; kk < 2; ++kk) {
        FragBF bk;
        const int kr = nt * 16 + l15;
        bk.h[0] = *(const v8bf*)&ldsK[cur][kr * LDK + kk * 32 + half * 8];
        bk.h[1] = *(const v8bf*)&ldsK[cur][kr * LDK + kk * 32 + 16 + half * 8];
        s_acc[nt] = WMMA_BF16(qa[kk].v, bk.v, s_acc[nt]);
      }
    }

    // scale+mask+online softmax (exp2 domain; row stats replicated per half)
    float f[8], mnew[8];
#pragma unroll
    for (int r = 0; r < 8; ++r) {
      const int qrow = qw + r + 8 * half;
      const unsigned* wp = &mbits[(size_t)qrow * (Ss / 32) + kt * 2];
      const unsigned w0 = wp[0], w1 = wp[1];       // 64 mask bits for this tile
      float mx = m_i[r];
#pragma unroll
      for (int nt = 0; nt < 4; ++nt) {
        const unsigned w = (nt < 2) ? w0 : w1;
        const int bit = (nt & 1) * 16 + l15;
        float sv = s_acc[nt][r] * kScaleLog2e;
        if (!((w >> bit) & 1u)) sv = -1e9f;
        s_acc[nt][r] = sv;
        mx = fmaxf(mx, sv);
      }
#pragma unroll
      for (int sh = 1; sh <= 8; sh <<= 1)
        mx = fmaxf(mx, __shfl_xor(mx, sh, 32));
      mnew[r] = mx;
      f[r] = exp2f(m_i[r] - mx);
    }

    float rs[8];
#pragma unroll
    for (int r = 0; r < 8; ++r) rs[r] = 0.f;
#pragma unroll
    for (int nt = 0; nt < 4; ++nt)
#pragma unroll
      for (int r = 0; r < 8; ++r) {
        const float p = exp2f(s_acc[nt][r] - mnew[r]);
        rs[r] += p;
        ldsP[wave][(r + 8 * half) * LDK + nt * 16 + l15] = (bf16_t)p;
      }
#pragma unroll
    for (int r = 0; r < 8; ++r) {
      float s = rs[r];
#pragma unroll
      for (int sh = 1; sh <= 8; sh <<= 1) s += __shfl_xor(s, sh, 32);
      l_i[r] = l_i[r] * f[r] + s;
      m_i[r] = mnew[r];
    }
#pragma unroll
    for (int nt = 0; nt < 4; ++nt)
#pragma unroll
      for (int r = 0; r < 8; ++r) o_acc[nt][r] *= f[r];

    // O += P @ V  (P re-read from LDS in A-layout; same-wave DS is in-order)
    FragBF pa[2];
#pragma unroll
    for (int kk = 0; kk < 2; ++kk) {
      pa[kk].h[0] = *(const v8bf*)&ldsP[wave][l15 * LDK + kk * 32 + half * 8];
      pa[kk].h[1] = *(const v8bf*)&ldsP[wave][l15 * LDK + kk * 32 + 16 + half * 8];
    }
#pragma unroll
    for (int nt = 0; nt < 4; ++nt) {
#pragma unroll
      for (int kk = 0; kk < 2; ++kk) {
        FragBF bv;
        const int hd = nt * 16 + l15;
        bv.h[0] = *(const v8bf*)&ldsV[cur][hd * LDK + kk * 32 + half * 8];
        bv.h[1] = *(const v8bf*)&ldsV[cur][hd * LDK + kk * 32 + 16 + half * 8];
        o_acc[nt] = WMMA_BF16(pa[kk].v, bv.v, o_acc[nt]);
      }
    }

    if (kt + 1 < NT) st_tiles(cur ^ 1);            // fill other buffer
    __syncthreads();
  }

  // normalize and write merged-head bf16 output [B,S,D]
#pragma unroll
  for (int nt = 0; nt < 4; ++nt)
#pragma unroll
    for (int r = 0; r < 8; ++r) {
      const int qrow = qw + r + 8 * half;
      const int hd = nt * 16 + l15;
      const float v = o_acc[nt][r] / l_i[r];
      Oa[((size_t)b * Ss + qrow) * Dd + h * HDd + hd] = (bf16_t)v;
    }
}

// ---------------------------------------------------------------------------
// Kernel 3: output projection. X bf16 [8192,1024] (attn out), W fp32,
// out fp32 row-major. Same double-buffered pipeline as kernel 1.
// ---------------------------------------------------------------------------
__global__ __launch_bounds__(256) void out_proj_kernel(
    const bf16_t* __restrict__ X, const float* __restrict__ W,
    float* __restrict__ Out)
{
  constexpr int BM = 128, BN = 128, BK = 32, LDT = BK + 8;
  constexpr int NK = Dd / BK;
  __shared__ bf16_t ldsA[2][BM * LDT];
  __shared__ bf16_t ldsB[2][BN * LDT];

  const int tid  = threadIdx.x;
  const int lane = tid & 31, wave = tid >> 5;
  const int l15  = lane & 15, half = lane >> 4;
  const int waveM = wave >> 1, waveN = wave & 1;
  const int m0 = blockIdx.y * BM;
  const int n0 = blockIdx.x * BN;
  const int ldRow = tid >> 1;
  const int ldCol = (tid & 1) * 16;

  const bf16_t* xrow = &X[(size_t)(m0 + ldRow) * Dd + ldCol];
  const float*  wrow = &W[(size_t)(n0 + ldRow) * Dd + ldCol];

  v8bf xa[2]; float4 fb[4];
  auto ld_global = [&](int k0) {
#pragma unroll
    for (int j = 0; j < 2; ++j) xa[j] = *(const v8bf*)&xrow[k0 + 8 * j];
#pragma unroll
    for (int i = 0; i < 4; ++i) fb[i] = *(const float4*)&wrow[k0 + 4 * i];
  };
  auto st_lds = [&](int buf) {
    bf16_t* A = &ldsA[buf][ldRow * LDT + ldCol];
    bf16_t* Bt = &ldsB[buf][ldRow * LDT + ldCol];
#pragma unroll
    for (int j = 0; j < 2; ++j) *(v8bf*)&A[8 * j] = xa[j];
#pragma unroll
    for (int i = 0; i < 4; ++i) {
      v4bf bb; bb[0] = (bf16_t)fb[i].x; bb[1] = (bf16_t)fb[i].y;
               bb[2] = (bf16_t)fb[i].z; bb[3] = (bf16_t)fb[i].w;
      *(v4bf*)&Bt[4 * i] = bb;
    }
  };

  const v8f vzero = {0.f,0.f,0.f,0.f,0.f,0.f,0.f,0.f};
  v8f acc[2][4];
#pragma unroll
  for (int mi = 0; mi < 2; ++mi)
#pragma unroll
    for (int ni = 0; ni < 4; ++ni) acc[mi][ni] = vzero;

  ld_global(0);
  st_lds(0);
  __syncthreads();

  for (int kt = 0; kt < NK; ++kt) {
    const int cur = kt & 1;
    if (kt + 1 < NK) ld_global((kt + 1) * BK);
    if (kt + 2 < NK) {
      __builtin_prefetch(&xrow[(kt + 2) * BK], 0, 1);
      __builtin_prefetch(&wrow[(kt + 2) * BK], 0, 1);
    }

    FragBF a[2], b[4];
#pragma unroll
    for (int mi = 0; mi < 2; ++mi) {
      const int r = waveM * 32 + mi * 16 + l15;
      a[mi].h[0] = *(const v8bf*)&ldsA[cur][r * LDT + half * 8];
      a[mi].h[1] = *(const v8bf*)&ldsA[cur][r * LDT + 16 + half * 8];
    }
#pragma unroll
    for (int ni = 0; ni < 4; ++ni) {
      const int c = waveN * 64 + ni * 16 + l15;
      b[ni].h[0] = *(const v8bf*)&ldsB[cur][c * LDT + half * 8];
      b[ni].h[1] = *(const v8bf*)&ldsB[cur][c * LDT + 16 + half * 8];
    }
#pragma unroll
    for (int ni = 0; ni < 4; ++ni)
#pragma unroll
      for (int mi = 0; mi < 2; ++mi)
        acc[mi][ni] = WMMA_BF16(a[mi].v, b[ni].v, acc[mi][ni]);

    if (kt + 1 < NK) st_lds(cur ^ 1);
    __syncthreads();
  }

#pragma unroll
  for (int mi = 0; mi < 2; ++mi)
#pragma unroll
    for (int ni = 0; ni < 4; ++ni)
#pragma unroll
      for (int r = 0; r < 8; ++r) {
        const int gm = m0 + waveM * 32 + mi * 16 + r + 8 * half;
        const int gn = n0 + waveN * 64 + ni * 16 + l15;
        Out[(size_t)gm * Dd + gn] = acc[mi][ni][r];
      }
}

// ---------------------------------------------------------------------------
extern "C" void kernel_launch(void* const* d_in, const int* in_sizes, int n_in,
                              void* d_out, int out_size, void* d_ws, size_t ws_size,
                              hipStream_t stream) {
  const float* query = (const float*)d_in[0];
  const float* key_  = (const float*)d_in[1];
  const float* value = (const float*)d_in[2];
  const int*   mask  = (const int*)d_in[3];
  const float* w_q   = (const float*)d_in[4];
  const float* w_k   = (const float*)d_in[5];
  const float* w_v   = (const float*)d_in[6];
  const float* w_o   = (const float*)d_in[7];
  float* out = (float*)d_out;

  const size_t elems = (size_t)Bb * Ss * Dd;       // 8,388,608
  bf16_t* Qh = (bf16_t*)d_ws;                      // [B,H,S,HD] bf16, 16 MB
  bf16_t* Kh = Qh + elems;                         // 16 MB
  bf16_t* Vh = Kh + elems;                         // 16 MB
  bf16_t* Ao = Vh + elems;                         // [B,S,D] bf16, 16 MB
  unsigned* mbits = (unsigned*)(Ao + elems);       // S*S/8 = 512 KB packed mask

  mask_pack_kernel<<<(Ss * Ss) / 256, 256, 0, stream>>>(mask, mbits);

  dim3 gProj(Dd / 128, Mtot / 128, 3);
  qkv_proj_kernel<<<gProj, 256, 0, stream>>>(query, key_, value, w_q, w_k, w_v,
                                             Qh, Kh, Vh);

  dim3 gAttn(Ss / 64, Bb * Hh);
  attn_kernel<<<gAttn, 128, 0, stream>>>(Qh, Kh, Vh, mbits, Ao);

  dim3 gO(Dd / 128, Mtot / 128, 1);
  out_proj_kernel<<<gO, 256, 0, stream>>>(Ao, w_o, out);
}
